// HANLayer_88888643158477
// MI455X (gfx1250) — matrix operand order repaired and checked by reading, not compiled, not run on hardware
//
#include <hip/hip_runtime.h>
#include <hip/hip_bf16.h>

// ---------------------------------------------------------------------------
// HAN layer for MI455X (gfx1250, wave32).
// fp32 GEMMs run on v_wmma_f32_16x16x32_bf16 via bf16 hi/lo split (3 WMMA per
// tile, "bf16x3" fp32 emulation). The hi/lo split is precomputed into global
// bf16 planes so the GEMM inner loop is pure ds_load_b128 + v_wmma. el/er
// attention dots are fused into the GEMM epilogue (register-resident acc +
// half-wave shfl_xor reduction). Edge softmax/scatter is L2-atomic bound.
// ---------------------------------------------------------------------------

typedef __attribute__((ext_vector_type(16))) __bf16 v16bf;
typedef __attribute__((ext_vector_type(8)))  __bf16 v8bf;
typedef __attribute__((ext_vector_type(8)))  float  v8f;

static constexpr int N_NODES = 50000;
static constexpr int E_EDGES = 800000;
static constexpr int P_PATHS = 3;
static constexpr int IN_F    = 256;
static constexpr int D_OUT   = 64;
static constexpr int SEM_HID = 128;
static constexpr float NEG_SLOPE = 0.2f;

// LDS tile stride (bf16 elements). 40*2B = 80B row pitch: keeps every
// fragment chunk 16B-aligned and staggers banks.
static constexpr int LDA = 40;

// ---------------- WMMA fragment loaders (ISA 7.12.2 layouts) ----------------
// A 16x32 bf16, row-major LDS tile: lane l -> row (l&15); K-chunks
// [half*8 .. half*8+7] and [16+half*8 .. 23+half*8]  (half = l>>4).
__device__ __forceinline__ v16bf load_a_frag(const __bf16* __restrict__ sA, int lane) {
  const int half = lane >> 4;
  const __bf16* row = sA + (lane & 15) * LDA;
  const v8bf c0 = *(const v8bf*)(row + half * 8);
  const v8bf c1 = *(const v8bf*)(row + 16 + half * 8);
  return __builtin_shufflevector(c0, c1, 0, 1, 2, 3, 4, 5, 6, 7,
                                 8, 9, 10, 11, 12, 13, 14, 15);
}

// B 32x16 bf16, column-major LDS tile: lane l -> col coloff+(l&15);
// K-chunks [half*8 .. half*8+7] and [16+half*8 .. 23+half*8] down the column.
__device__ __forceinline__ v16bf load_b_frag(const __bf16* __restrict__ sB, int coloff,
                                             int lane) {
  const int half = lane >> 4;
  const __bf16* col = sB + (coloff + (lane & 15)) * LDA;
  const v8bf c0 = *(const v8bf*)(col + half * 8);
  const v8bf c1 = *(const v8bf*)(col + 16 + half * 8);
  return __builtin_shufflevector(c0, c1, 0, 1, 2, 3, 4, 5, 6, 7,
                                 8, 9, 10, 11, 12, 13, 14, 15);
}

// acc += (Ah+Al)(Bh+Bl), dropping lo*lo (bf16x3 fp32 emulation).
__device__ __forceinline__ v8f wmma_bf16x3(v16bf ah, v16bf al, v16bf bh, v16bf bl, v8f c) {
  c = __builtin_amdgcn_wmma_f32_16x16x32_bf16(false, al, false, bh, (short)0, c, false, false);
  c = __builtin_amdgcn_wmma_f32_16x16x32_bf16(false, ah, false, bl, (short)0, c, false, false);
  c = __builtin_amdgcn_wmma_f32_16x16x32_bf16(false, ah, false, bh, (short)0, c, false, false);
  return c;
}

// ------------------------- kernel 0: fills ----------------------------------
__global__ void fill_u32_kernel(unsigned* __restrict__ p, unsigned val, size_t n) {
  size_t i = (size_t)blockIdx.x * blockDim.x + threadIdx.x;
  if (i < n) p[i] = val;
}

// ------------- kernel 0b: fp32 -> bf16 hi/lo split planes -------------------
__global__ __launch_bounds__(256) void split_bf16_kernel(const float* __restrict__ in,
                                                         __bf16* __restrict__ hi,
                                                         __bf16* __restrict__ lo, size_t n) {
  const size_t i = (size_t)blockIdx.x * blockDim.x + threadIdx.x;
  if (i >= n) return;
  const float v = in[i];
  const __bf16 h = (__bf16)v;
  hi[i] = h;
  lo[i] = (__bf16)(v - (float)h);
}

// ------- kernel 1: feat[p] = h @ W[p] (WMMA bf16x3) + fused el/er -----------
// Block = 256 threads (8 waves). Each block: 128 rows x 64 cols for path p.
__global__ __launch_bounds__(256) void feat_gemm_kernel(const __bf16* __restrict__ h_hi,
                                                        const __bf16* __restrict__ h_lo,
                                                        const __bf16* __restrict__ W_hi,
                                                        const __bf16* __restrict__ W_lo,
                                                        const float* __restrict__ attn_l,
                                                        const float* __restrict__ attn_r,
                                                        float* __restrict__ feat,
                                                        float* __restrict__ el,
                                                        float* __restrict__ er) {
  const int p    = blockIdx.y;
  const int row0 = blockIdx.x * 128;
  __shared__ __align__(16) __bf16 sAh[128 * LDA], sAl[128 * LDA];  // rows x k
  __shared__ __align__(16) __bf16 sBh[64 * LDA],  sBl[64 * LDA];   // cols x k (col-major)
  const int tid  = threadIdx.x;
  const int wave = tid >> 5;
  const int lane = tid & 31;
  const int half = lane >> 4;

  v8f acc[4] = {};

  for (int k0 = 0; k0 < IN_F; k0 += 32) {
    // A tiles: 128 rows x 32 k as bf16-pairs (uint), 2048 uints per plane
    for (int i = tid; i < 128 * 16; i += 256) {
      const int r = i >> 4, cp = i & 15;
      const int row = row0 + r;
      unsigned vh = 0u, vl = 0u;
      if (row < N_NODES) {
        const size_t base = ((size_t)row * IN_F + k0) / 2 + cp;
        vh = ((const unsigned*)h_hi)[base];
        vl = ((const unsigned*)h_lo)[base];
      }
      ((unsigned*)sAh)[r * (LDA / 2) + cp] = vh;
      ((unsigned*)sAl)[r * (LDA / 2) + cp] = vl;
    }
    // B tiles: 32 k x 64 n, transposed into column-major LDS
    for (int i = tid; i < 32 * 64; i += 256) {
      const int r = i >> 6, c = i & 63;
      const size_t g = ((size_t)p * IN_F + k0 + r) * D_OUT + c;
      sBh[c * LDA + r] = W_hi[g];
      sBl[c * LDA + r] = W_lo[g];
    }
    __syncthreads();

    const v16bf ah = load_a_frag(sAh + wave * 16 * LDA, lane);
    const v16bf al = load_a_frag(sAl + wave * 16 * LDA, lane);
#pragma unroll
    for (int t = 0; t < 4; ++t) {
      const v16bf bh = load_b_frag(sBh, t * 16, lane);
      const v16bf bl = load_b_frag(sBl, t * 16, lane);
      acc[t] = wmma_bf16x3(ah, al, bh, bl, acc[t]);
    }
    __syncthreads();
  }

  // ---- fused el/er: per-lane partial dots over this lane's columns ----
  const int nloc = lane & 15;
  float pl[8], pr[8];
#pragma unroll
  for (int j = 0; j < 8; ++j) { pl[j] = 0.f; pr[j] = 0.f; }
#pragma unroll
  for (int t = 0; t < 4; ++t) {
    const int col = t * 16 + nloc;
    const float cl = attn_l[p * D_OUT + col];
    const float cr = attn_r[p * D_OUT + col];
#pragma unroll
    for (int j = 0; j < 8; ++j) {
      pl[j] += acc[t][j] * cl;
      pr[j] += acc[t][j] * cr;
    }
  }
#pragma unroll
  for (int msk = 1; msk < 16; msk <<= 1) {
#pragma unroll
    for (int j = 0; j < 8; ++j) {
      pl[j] += __shfl_xor(pl[j], msk, 32);
      pr[j] += __shfl_xor(pr[j], msk, 32);
    }
  }

  // ---- writeback: one base pointer per lane, immediate offsets ----
  const int rbase = row0 + wave * 16 + half * 8;          // this lane's first row
  float* fb = feat + ((size_t)p * N_NODES + rbase) * D_OUT + nloc;
  float* elb = el + (size_t)p * N_NODES + rbase;
  float* erb = er + (size_t)p * N_NODES + rbase;
  if (row0 + 128 <= N_NODES) {  // full tile (all but the last row-block)
#pragma unroll
    for (int t = 0; t < 4; ++t)
#pragma unroll
      for (int j = 0; j < 8; ++j)
        fb[j * D_OUT + t * 16] = acc[t][j];
    if (nloc == 0) {
#pragma unroll
      for (int j = 0; j < 8; ++j) { elb[j] = pl[j]; erb[j] = pr[j]; }
    }
  } else {
#pragma unroll
    for (int j = 0; j < 8; ++j) {
      if (rbase + j < N_NODES) {
#pragma unroll
        for (int t = 0; t < 4; ++t) fb[j * D_OUT + t * 16] = acc[t][j];
        if (nloc == 0) { elb[j] = pl[j]; erb[j] = pr[j]; }
      }
    }
  }
}

// ----- order-preserving float<->uint encoding for atomic segment-max --------
__device__ __forceinline__ unsigned enc_f32(float v) {
  unsigned b = __float_as_uint(v);
  return (b & 0x80000000u) ? ~b : (b | 0x80000000u);
}
__device__ __forceinline__ float dec_f32(unsigned e) {
  return (e & 0x80000000u) ? __uint_as_float(e & 0x7fffffffu) : __uint_as_float(~e);
}

// ------------- kernel 3: per-edge leaky score, segment max ------------------
__global__ __launch_bounds__(256) void edge_max_kernel(const int* __restrict__ src,
                                                       const int* __restrict__ dst,
                                                       const float* __restrict__ el,
                                                       const float* __restrict__ er,
                                                       unsigned* __restrict__ mbuf) {
  const int e = blockIdx.x * blockDim.x + threadIdx.x;
  if (e >= E_EDGES) return;
  const int s = src[e], d = dst[e];
  float v = el[s] + er[d];
  v = (v > 0.f) ? v : NEG_SLOPE * v;
  atomicMax(&mbuf[d], enc_f32(v));
}

// ------------- kernel 4: softmax numerator + 64-wide scatter-add ------------
// One wave per edge; 32 lanes cover 64 feature columns (2 each).
__global__ __launch_bounds__(256) void edge_scatter_kernel(const int* __restrict__ src,
                                                           const int* __restrict__ dst,
                                                           const float* __restrict__ el,
                                                           const float* __restrict__ er,
                                                           const unsigned* __restrict__ mbuf,
                                                           const float* __restrict__ feat,
                                                           float* __restrict__ denom,
                                                           float* __restrict__ agg) {
  const int wave = (blockIdx.x * blockDim.x + threadIdx.x) >> 5;
  const int lane = threadIdx.x & 31;
  if (wave >= E_EDGES) return;
  const int s = src[wave], d = dst[wave];
  float v = el[s] + er[d];
  v = (v > 0.f) ? v : NEG_SLOPE * v;
  float m = dec_f32(mbuf[d]);
  if (!(m > -1e30f)) m = 0.f;  // empty-segment guard (mirrors reference)
  const float ex = __expf(v - m);
  if (lane == 0) atomicAdd(&denom[d], ex);
  const float* fs = feat + (size_t)s * D_OUT;
  float*       ad = agg + (size_t)d * D_OUT;
  atomicAdd(&ad[lane],      ex * fs[lane]);
  atomicAdd(&ad[lane + 32], ex * fs[lane + 32]);
}

// ------- kernel 5: z = elu(agg/denom) in place + fused bf16 hi/lo split -----
__global__ __launch_bounds__(256) void elu_split_kernel(const float* __restrict__ denom,
                                                        float* __restrict__ z,
                                                        __bf16* __restrict__ z_hi,
                                                        __bf16* __restrict__ z_lo) {
  const size_t i = (size_t)blockIdx.x * blockDim.x + threadIdx.x;  // over P*N*64
  if (i >= (size_t)P_PATHS * N_NODES * D_OUT) return;
  const float dn = fmaxf(denom[i >> 6], 1e-9f);
  float v = z[i] / dn;
  v = (v > 0.f) ? v : (__expf(v) - 1.f);
  z[i] = v;
  const __bf16 h = (__bf16)v;
  z_hi[i] = h;
  z_lo[i] = (__bf16)(v - (float)h);
}

// ------ kernel 6: semantic scores via WMMA: w = tanh(z@W1+b1)@w2 ------------
// Rows = P*N = 150000 as one GEMM; per-row score reduced in-register and
// accumulated into per-path sums (block-level LDS reduction first).
__global__ __launch_bounds__(256) void sem_gemm_kernel(const __bf16* __restrict__ z_hi,
                                                       const __bf16* __restrict__ z_lo,
                                                       const __bf16* __restrict__ W1_hi,
                                                       const __bf16* __restrict__ W1_lo,
                                                       const float* __restrict__ b1,
                                                       const float* __restrict__ w2,
                                                       float* __restrict__ wsum) {
  const int row0 = blockIdx.x * 128;
  __shared__ __align__(16) __bf16 sAh[128 * LDA], sAl[128 * LDA];
  __shared__ __align__(16) __bf16 sBh[SEM_HID * LDA], sBl[SEM_HID * LDA];
  __shared__ float sWsum[P_PATHS];
  const int tid  = threadIdx.x;
  const int wave = tid >> 5;
  const int lane = tid & 31;
  const int half = lane >> 4;
  if (tid < P_PATHS) sWsum[tid] = 0.f;

  v8f acc[8] = {};

  for (int k0 = 0; k0 < D_OUT; k0 += 32) {
    for (int i = tid; i < 128 * 16; i += 256) {
      const int r = i >> 4, cp = i & 15;
      const int row = row0 + r;
      unsigned vh = 0u, vl = 0u;
      if (row < P_PATHS * N_NODES) {
        const size_t base = ((size_t)row * D_OUT + k0) / 2 + cp;
        vh = ((const unsigned*)z_hi)[base];
        vl = ((const unsigned*)z_lo)[base];
      }
      ((unsigned*)sAh)[r * (LDA / 2) + cp] = vh;
      ((unsigned*)sAl)[r * (LDA / 2) + cp] = vl;
    }
    for (int i = tid; i < 32 * SEM_HID; i += 256) {
      const int r = i >> 7, c = i & 127;
      const size_t g = (size_t)(k0 + r) * SEM_HID + c;
      sBh[c * LDA + r] = W1_hi[g];
      sBl[c * LDA + r] = W1_lo[g];
    }
    __syncthreads();

    const v16bf ah = load_a_frag(sAh + wave * 16 * LDA, lane);
    const v16bf al = load_a_frag(sAl + wave * 16 * LDA, lane);
#pragma unroll
    for (int t = 0; t < 8; ++t) {
      const v16bf bh = load_b_frag(sBh, t * 16, lane);
      const v16bf bl = load_b_frag(sBl, t * 16, lane);
      acc[t] = wmma_bf16x3(ah, al, bh, bl, acc[t]);
    }
    __syncthreads();
  }

  // per-lane partial of sum_c tanh(acc+b1[c]) * w2[c] over this lane's columns
  const int nloc = lane & 15;
  float part[8];
#pragma unroll
  for (int j = 0; j < 8; ++j) part[j] = 0.f;
#pragma unroll
  for (int t = 0; t < 8; ++t) {
    const int col = t * 16 + nloc;
    const float bb = b1[col];
    const float ww = w2[col];
#pragma unroll
    for (int j = 0; j < 8; ++j) part[j] += tanhf(acc[t][j] + bb) * ww;
  }
  // reduce across the 16 lanes of each half-wave (xor masks stay in-half)
#pragma unroll
  for (int msk = 1; msk < 16; msk <<= 1) {
#pragma unroll
    for (int j = 0; j < 8; ++j) part[j] += __shfl_xor(part[j], msk, 32);
  }
  if (nloc == 0) {
#pragma unroll
    for (int j = 0; j < 8; ++j) {
      const int row = row0 + wave * 16 + j + half * 8;
      if (row < P_PATHS * N_NODES) atomicAdd(&sWsum[row / N_NODES], part[j]);
    }
  }
  __syncthreads();
  if (tid < P_PATHS) atomicAdd(&wsum[tid], sWsum[tid]);
}

// ------------- kernel 7: beta = softmax(mean_n w) ---------------------------
__global__ void beta_kernel(const float* __restrict__ wsum, float* __restrict__ beta) {
  if (threadIdx.x == 0) {
    const float inv = 1.f / (float)N_NODES;
    const float w0 = wsum[0] * inv, w1 = wsum[1] * inv, w2 = wsum[2] * inv;
    const float mx = fmaxf(w0, fmaxf(w1, w2));
    const float e0 = __expf(w0 - mx), e1 = __expf(w1 - mx), e2 = __expf(w2 - mx);
    const float s = e0 + e1 + e2;
    beta[0] = e0 / s;
    beta[1] = e1 / s;
    beta[2] = e2 / s;
  }
}

// ------------- kernel 8: out = sum_p beta[p] * z[p] -------------------------
__global__ __launch_bounds__(256) void combine_kernel(const float* __restrict__ z,
                                                      const float* __restrict__ beta,
                                                      float* __restrict__ out) {
  const size_t i = (size_t)blockIdx.x * blockDim.x + threadIdx.x;  // over N*64
  if (i >= (size_t)N_NODES * D_OUT) return;
  const size_t stride = (size_t)N_NODES * D_OUT;
  out[i] = beta[0] * z[i] + beta[1] * z[stride + i] + beta[2] * z[2 * stride + i];
}

// ---------------------------------------------------------------------------
extern "C" void kernel_launch(void* const* d_in, const int* in_sizes, int n_in,
                              void* d_out, int out_size, void* d_ws, size_t ws_size,
                              hipStream_t stream) {
  (void)in_sizes; (void)n_in; (void)out_size; (void)ws_size;

  const float* h      = (const float*)d_in[0];
  const int*   srcs[3] = {(const int*)d_in[1], (const int*)d_in[3], (const int*)d_in[5]};
  const int*   dsts[3] = {(const int*)d_in[2], (const int*)d_in[4], (const int*)d_in[6]};
  const float* W      = (const float*)d_in[7];
  const float* attn_l = (const float*)d_in[8];
  const float* attn_r = (const float*)d_in[9];
  const float* sem_W1 = (const float*)d_in[10];
  const float* sem_b1 = (const float*)d_in[11];
  const float* sem_w2 = (const float*)d_in[12];
  float* out = (float*)d_out;

  // workspace layout (~169 MB)
  const size_t PN   = (size_t)P_PATHS * N_NODES;
  const size_t PND  = PN * D_OUT;              // 9.6M
  const size_t NH   = (size_t)N_NODES * IN_F;  // 12.8M
  const size_t NW   = (size_t)P_PATHS * IN_F * D_OUT;
  const size_t NW1  = (size_t)D_OUT * SEM_HID;
  float*    feat  = (float*)d_ws;              // PND f32
  float*    zbuf  = feat + PND;                // PND f32 (agg, then z in place)
  float*    el    = zbuf + PND;                // PN
  float*    er    = el + PN;                   // PN
  unsigned* mbuf  = (unsigned*)(er + PN);      // PN
  float*    denom = (float*)(mbuf + PN);       // PN
  float*    wsum  = denom + PN;                // 8 (3 used)
  float*    beta  = wsum + 8;                  // 8 (3 used)
  __bf16*   h_hi  = (__bf16*)(beta + 8);       // NH
  __bf16*   h_lo  = h_hi + NH;                 // NH
  __bf16*   z_hi  = h_lo + NH;                 // PND
  __bf16*   z_lo  = z_hi + PND;                // PND
  __bf16*   W_hi  = z_lo + PND;                // NW
  __bf16*   W_lo  = W_hi + NW;                 // NW
  __bf16*   W1_hi = W_lo + NW;                 // NW1
  __bf16*   W1_lo = W1_hi + NW1;               // NW1

  // 0) init accumulators (every call; workspace is not re-poisoned between replays)
  fill_u32_kernel<<<(unsigned)((PND + 255) / 256), 256, 0, stream>>>((unsigned*)zbuf, 0u, PND);
  fill_u32_kernel<<<(unsigned)((PN + 8 + 255) / 256), 256, 0, stream>>>((unsigned*)denom, 0u, PN + 8);
  fill_u32_kernel<<<(unsigned)((PN + 255) / 256), 256, 0, stream>>>(mbuf, 0x007FFFFFu /* enc(-inf) */, PN);

  // 0b) bf16 hi/lo planes for WMMA operands
  split_bf16_kernel<<<(unsigned)((NH + 255) / 256), 256, 0, stream>>>(h, h_hi, h_lo, NH);
  split_bf16_kernel<<<(unsigned)((NW + 255) / 256), 256, 0, stream>>>(W, W_hi, W_lo, NW);
  split_bf16_kernel<<<(unsigned)((NW1 + 255) / 256), 256, 0, stream>>>(sem_W1, W1_hi, W1_lo, NW1);

  // 1) feat[p] = h @ W[p]   (WMMA) + fused el/er
  feat_gemm_kernel<<<dim3((N_NODES + 127) / 128, P_PATHS), 256, 0, stream>>>(
      h_hi, h_lo, W_hi, W_lo, attn_l, attn_r, feat, el, er);

  // 2) segment max per path
  for (int p = 0; p < P_PATHS; ++p)
    edge_max_kernel<<<E_EDGES / 256, 256, 0, stream>>>(srcs[p], dsts[p],
        el + (size_t)p * N_NODES, er + (size_t)p * N_NODES, mbuf + (size_t)p * N_NODES);

  // 3) softmax numerator + scatter-add (one wave per edge)
  for (int p = 0; p < P_PATHS; ++p)
    edge_scatter_kernel<<<E_EDGES * 32 / 256, 256, 0, stream>>>(srcs[p], dsts[p],
        el + (size_t)p * N_NODES, er + (size_t)p * N_NODES, mbuf + (size_t)p * N_NODES,
        feat + (size_t)p * N_NODES * D_OUT, denom + (size_t)p * N_NODES,
        zbuf + (size_t)p * N_NODES * D_OUT);

  // 4) z = elu(agg/denom), fused bf16 split for the semantic GEMM
  elu_split_kernel<<<(unsigned)((PND + 255) / 256), 256, 0, stream>>>(denom, zbuf, z_hi, z_lo);

  // 5) semantic attention scores (WMMA) -> per-path sums
  sem_gemm_kernel<<<(int)((PN + 127) / 128), 256, 0, stream>>>(
      z_hi, z_lo, W1_hi, W1_lo, sem_b1, sem_w2, wsum);

  // 6) beta = softmax(mean)
  beta_kernel<<<1, 32, 0, stream>>>(wsum, beta);

  // 7) out = sum_p beta[p] * z[p]
  combine_kernel<<<(unsigned)(((size_t)N_NODES * D_OUT + 255) / 256), 256, 0, stream>>>(zbuf, beta, out);
}
